// BiLSTMSeq2Seq_3899830304990
// MI455X (gfx1250) — matrix operand order repaired and checked by reading, not compiled, not run on hardware
//
#include <hip/hip_runtime.h>
#include <cstdint>

// ---------------------------------------------------------------------------
// BiLSTM seq2seq + Bahdanau attention on CDNA5 (gfx1250).
//  - all GEMMs through v_wmma_f32_16x16x32_bf16 (wave32), fp32 accumulate
//  - attention concat matmul split: Wa@[h;enc] = Wa_h@h + Wa_e@enc (hoisted)
//  - big GEMMs: block-cooperative, B panel staged in LDS (8x reuse across
//    the block's 8 M-tiles), filled by the Tensor Data Mover when available
//    (double-buffered, TENSORcnt-synchronized), else cooperative copy
//  - recurrent steps: fused dual-direction / dual-product WMMA kernels
// ---------------------------------------------------------------------------

typedef __bf16 bf16_t;
typedef __attribute__((ext_vector_type(16))) __bf16   v16bf;
typedef __attribute__((ext_vector_type(8)))  __bf16   v8bf;
typedef __attribute__((ext_vector_type(8)))  float    v8f;
typedef __attribute__((ext_vector_type(4)))  uint32_t u32x4;
typedef __attribute__((ext_vector_type(4)))  int      i32x4;
typedef __attribute__((ext_vector_type(8)))  int      i32x8;

#define BB    32
#define TSRC  64
#define TTGT  64
#define EDIM  512
#define HDIM  512
#define H2    1024
#define H4    2048
#define H8    4096
#define VOUT  32000

// Big-GEMM LDS staging geometry: 64 N-rows x 128 K-elems per chunk,
// row padded 256B->272B so ds_load_b128 fragment reads are conflict-free.
#define KCHUNK       128
#define NROWS        64
#define LDS_ROW_B    272
#define LDS_BUF_B    (NROWS * LDS_ROW_B)      // 17408 B per buffer

#if __has_builtin(__builtin_amdgcn_tensor_load_to_lds) && __has_builtin(__builtin_amdgcn_s_wait_tensorcnt)
#define HAVE_TDM 1
#if __has_include(<hip/amd_detail/amd_gfx1250_TDM.h>)
#define TDM_6ARG 1
#endif
#endif

__device__ __forceinline__ float sigmoidf_(float x) { return 1.0f / (1.0f + __expf(-x)); }

// --- WMMA fragment loaders (contiguous k = 16*khalf + s, same order for A and B) ----------------------------
// A (16-bit 16x32): lanes 0-15: elems K0..15; lanes 16-31: K16..31
__device__ __forceinline__ v16bf load_fragA(const bf16_t* rowptr, int k0, int khalf)
{
    union { v16bf v; v8bf h[2]; } f;
    f.h[0] = *(const v8bf*)(rowptr + k0 + khalf * 16);
    f.h[1] = *(const v8bf*)(rowptr + k0 + khalf * 16 + 8);
    return f.v;
}
// B (32x16): lane holds col N=lane&15; 16 contiguous K at k0 + (lane>=16)*16
__device__ __forceinline__ v16bf load_fragB(const bf16_t* W, int ldw, int nrow, int k0, int khalf)
{
    const bf16_t* p = W + (size_t)nrow * ldw + k0 + khalf * 16;
    union { v16bf v; v8bf h[2]; } f;
    f.h[0] = *(const v8bf*)p;
    f.h[1] = *(const v8bf*)(p + 8);
    return f.v;
}
__device__ __forceinline__ v16bf load_fragB_lds(const char* buf, int trow, int kk, int khalf)
{
    const bf16_t* p = (const bf16_t*)(buf + trow * LDS_ROW_B) + kk + khalf * 16;
    union { v16bf v; v8bf h[2]; } f;
    f.h[0] = *(const v8bf*)p;
    f.h[1] = *(const v8bf*)(p + 8);
    return f.v;
}
#define WMMA_BF16(a, b, c) \
    __builtin_amdgcn_wmma_f32_16x16x32_bf16(false, (a), false, (b), (short)0, (c), false, false)

#ifdef HAVE_TDM
__device__ __forceinline__ uint32_t lds_addr_of(const void* p)
{
    return (uint32_t)(uintptr_t)(__attribute__((address_space(3))) const char*)p;
}
// TDM 2D tile load: tile_k x tile_n bf16 tile from W (row stride ldw elems)
// into LDS with pad_amount=4dw after every 64dw (one row) -> 272B rows.
__device__ __forceinline__ void tdm_load_tile(uint32_t lds_off, const bf16_t* gptr, uint32_t ldw)
{
    uint64_t ga = (uint64_t)(uintptr_t)gptr;
    u32x4 g0 = { 1u,                                       // count=1 (valid user D#)
                 lds_off,                                  // lds_addr
                 (uint32_t)ga,                             // global_addr[31:0]
                 (uint32_t)((ga >> 32) & 0x01FFFFFFu) | (2u << 30) };  // addr[56:32] | type=2
    const uint32_t data_size = 1u;                         // 2-byte elements
    const uint32_t pad_int   = 5u;                         // 2^(5+1) = 64 dwords interval
    const uint32_t pad_amt   = 3u;                         // 4 dwords of pad
    uint32_t tdim0 = ldw;                                  // tensor dim0 (row length)
    uint32_t tdim1 = 1u << 20;                             // plenty of rows (no OOB clip)
    uint32_t w0 = (data_size << 16) | (1u << 20) | (pad_int << 22) | (pad_amt << 25);
    uint32_t w1 = (tdim0 & 0xFFFFu) << 16;                 // tensor_dim0[15:0] @ bits 63:48
    uint32_t w2 = (tdim0 >> 16) | ((tdim1 & 0xFFFFu) << 16);
    uint32_t w3 = (tdim1 >> 16) | ((uint32_t)KCHUNK << 16);// tile_dim0 @ bits 127:112
    uint32_t w4 = NROWS;                                   // tile_dim1 (tile_dim2 = 0)
    uint32_t w5 = ldw;                                     // tensor_dim0_stride[31:0]
    i32x8 g1 = { (int)w0, (int)w1, (int)w2, (int)w3, (int)w4, (int)w5, 0, 0 };
    i32x4 z4 = { 0, 0, 0, 0 };
#ifdef TDM_6ARG
    i32x8 z8 = { 0, 0, 0, 0, 0, 0, 0, 0 };
    __builtin_amdgcn_tensor_load_to_lds(g0, g1, z4, z4, z8, 0);
#else
    __builtin_amdgcn_tensor_load_to_lds(g0, g1, z4, z4, 0);
#endif
}
#endif // HAVE_TDM

// ---------------------------------------------------------------------------
// Block-cooperative big GEMM: C[M,N] = A[M,K] @ W[N,K]^T (+addC)(+bias)(relu)
// Block = 8 waves; wave w owns M-tile (blockIdx.y*128 + w*16); all waves share
// the block's 64-col B panel (4 N-tiles per wave, TN=4) staged in LDS.
// Requires M%128==0, N%64==0, K%KCHUNK==0.
// ---------------------------------------------------------------------------
__global__ __launch_bounds__(256) void gemm_big_kernel(
    float* __restrict__ C, const bf16_t* __restrict__ A, const bf16_t* __restrict__ W,
    const float* __restrict__ addC, const float* __restrict__ bias,
    int M, int N, int K, int lda, int ldw, int ldc, int relu)
{
    __shared__ char ldsB[2 * LDS_BUF_B];

    const int wave  = threadIdx.x >> 5;
    const int lane  = threadIdx.x & 31;
    const int n0    = blockIdx.x * NROWS;                 // block's 64-col panel
    const int m0    = blockIdx.y * 128 + wave * 16;       // wave's M tile
    const int khalf = lane >> 4;
    const int l15   = lane & 15;
    const int nch   = K / KCHUNK;

    v8f acc[4];
#pragma unroll
    for (int t = 0; t < 4; ++t)
#pragma unroll
        for (int r = 0; r < 8; ++r) acc[t][r] = 0.0f;

    const bf16_t* arow = A + (size_t)(m0 + l15) * lda;

#ifdef HAVE_TDM
    const uint32_t lds0 = lds_addr_of(ldsB);
    if (threadIdx.x == 0)
        tdm_load_tile(lds0, W + (size_t)n0 * ldw, (uint32_t)ldw);
    for (int c = 0; c < nch; ++c) {
        const int cur = c & 1;
        __syncthreads();                                  // everyone done reading buf[cur^1]
        if (threadIdx.x == 0) {
            if (c + 1 < nch) {
                tdm_load_tile(lds0 + (uint32_t)((cur ^ 1) * LDS_BUF_B),
                              W + (size_t)n0 * ldw + (size_t)(c + 1) * KCHUNK, (uint32_t)ldw);
                __builtin_amdgcn_s_wait_tensorcnt((short)1);  // fill(cur) done, next in flight
            } else {
                __builtin_amdgcn_s_wait_tensorcnt((short)0);
            }
        }
        __syncthreads();                                  // buf[cur] valid for all waves
        const char* buf = ldsB + cur * LDS_BUF_B;
        const int kc = c * KCHUNK;
        __builtin_prefetch(arow + kc + KCHUNK, 0, 1);     // global_prefetch for A stream
#pragma unroll
        for (int kk = 0; kk < KCHUNK; kk += 32) {
            v16bf af = load_fragA(arow, kc + kk, khalf);
#pragma unroll
            for (int t = 0; t < 4; ++t) {
                v16bf bf = load_fragB_lds(buf, t * 16 + l15, kk, khalf);
                acc[t] = WMMA_BF16(af, bf, acc[t]);
            }
        }
    }
#else
    for (int c = 0; c < nch; ++c) {
        const int kc = c * KCHUNK;
        {   // cooperative fill: 256 threads x 64B each row-quarter
            const int row = threadIdx.x >> 2, q = threadIdx.x & 3;
            const bf16_t* src = W + (size_t)(n0 + row) * ldw + kc + q * 32;
            char* dst = ldsB + row * LDS_ROW_B + q * 64;
#pragma unroll
            for (int i = 0; i < 4; ++i)
                *(v8bf*)(dst + i * 16) = *(const v8bf*)(src + i * 8);
        }
        __syncthreads();
        __builtin_prefetch(arow + kc + KCHUNK, 0, 1);
#pragma unroll
        for (int kk = 0; kk < KCHUNK; kk += 32) {
            v16bf af = load_fragA(arow, kc + kk, khalf);
#pragma unroll
            for (int t = 0; t < 4; ++t) {
                v16bf bf = load_fragB_lds(ldsB, t * 16 + l15, kk, khalf);
                acc[t] = WMMA_BF16(af, bf, acc[t]);
            }
        }
        __syncthreads();
    }
#endif

#pragma unroll
    for (int t = 0; t < 4; ++t) {
        const int col = n0 + t * 16 + l15;
        const float bv = bias ? bias[col] : 0.0f;
#pragma unroll
        for (int r = 0; r < 8; ++r) {
            const int row = m0 + khalf * 8 + r;
            const size_t off = (size_t)row * ldc + col;
            float v = acc[t][r] + bv + (addC ? addC[off] : 0.0f);
            if (relu) v = v > 0.0f ? v : 0.0f;
            C[off] = v;
        }
    }
}

// ---------------------------------------------------------------------------
// Small generic GEMM (TN=1) — used for Hq = h @ Wa_h.T per decoder step
// ---------------------------------------------------------------------------
__global__ __launch_bounds__(256) void gemm_small_kernel(
    float* __restrict__ C, const bf16_t* __restrict__ A, const bf16_t* __restrict__ W,
    int M, int N, int K, int lda, int ldw, int ldc)
{
    const int wave = threadIdx.x >> 5, lane = threadIdx.x & 31;
    const int tile = blockIdx.x * 8 + wave;
    const int n0 = tile * 16;
    if (n0 >= N) return;
    const int m0 = blockIdx.y * 16;
    const int khalf = lane >> 4, l15 = lane & 15;

    v8f acc;
#pragma unroll
    for (int r = 0; r < 8; ++r) acc[r] = 0.0f;
    const bf16_t* arow = A + (size_t)(m0 + l15) * lda;
    for (int k0 = 0; k0 < K; k0 += 32) {
        v16bf af = load_fragA(arow, k0, khalf);
        v16bf bf = load_fragB(W, ldw, n0 + l15, k0, khalf);
        acc = WMMA_BF16(af, bf, acc);
    }
#pragma unroll
    for (int r = 0; r < 8; ++r)
        C[(size_t)(m0 + khalf * 8 + r) * ldc + n0 + l15] = acc[r];
}

// ---------------------------------------------------------------------------
// Encoder recurrent step, both directions in one launch (blockIdx.z = dir):
//   G{f,b}[32,2048] = X{f,b}_t + h{f,b} @ Whh_{f,b}.T      (K = 512)
// ---------------------------------------------------------------------------
__global__ __launch_bounds__(256) void enc_step_kernel(
    float* __restrict__ Gf, float* __restrict__ Gb,
    const bf16_t* __restrict__ hf, const bf16_t* __restrict__ hb,
    const bf16_t* __restrict__ Whf, const bf16_t* __restrict__ Whb,
    const float* __restrict__ Xft, const float* __restrict__ Xbt)
{
    const int dir = blockIdx.z;
    float* C = dir ? Gb : Gf;
    const bf16_t* A = dir ? hb : hf;
    const bf16_t* W = dir ? Whb : Whf;
    const float* addC = dir ? Xbt : Xft;

    const int wave = threadIdx.x >> 5, lane = threadIdx.x & 31;
    const int n0 = (blockIdx.x * 8 + wave) * 16;          // N = 2048
    if (n0 >= H4) return;
    const int m0 = blockIdx.y * 16;                       // M = 32
    const int khalf = lane >> 4, l15 = lane & 15;

    v8f acc;
#pragma unroll
    for (int r = 0; r < 8; ++r) acc[r] = 0.0f;
    const bf16_t* arow = A + (size_t)(m0 + l15) * HDIM;
#pragma unroll
    for (int k0 = 0; k0 < HDIM; k0 += 32) {
        v16bf af = load_fragA(arow, k0, khalf);
        v16bf bf = load_fragB(W, HDIM, n0 + l15, k0, khalf);
        acc = WMMA_BF16(af, bf, acc);
    }
#pragma unroll
    for (int r = 0; r < 8; ++r) {
        const size_t off = (size_t)(m0 + khalf * 8 + r) * H4 + n0 + l15;
        C[off] = acc[r] + addC[off];
    }
}

// ---------------------------------------------------------------------------
// Decoder recurrent step: dual-product fused GEMM
//   Gd[32,4096] = Xd_t + cvec @ Wih_dc.T + h @ Whh_d.T   (both K = 1024)
// ---------------------------------------------------------------------------
__global__ __launch_bounds__(256) void dec_step_kernel(
    float* __restrict__ Gd, const bf16_t* __restrict__ cvec, const bf16_t* __restrict__ Wic,
    const bf16_t* __restrict__ h, const bf16_t* __restrict__ Whh,
    const float* __restrict__ Xdt)
{
    const int wave = threadIdx.x >> 5, lane = threadIdx.x & 31;
    const int n0 = (blockIdx.x * 8 + wave) * 16;          // N = 4096
    if (n0 >= H8) return;
    const int m0 = blockIdx.y * 16;                       // M = 32
    const int khalf = lane >> 4, l15 = lane & 15;

    v8f acc;
#pragma unroll
    for (int r = 0; r < 8; ++r) acc[r] = 0.0f;
    const bf16_t* arow0 = cvec + (size_t)(m0 + l15) * H2;
    const bf16_t* arow1 = h    + (size_t)(m0 + l15) * H2;
    for (int k0 = 0; k0 < H2; k0 += 32) {
        v16bf a0 = load_fragA(arow0, k0, khalf);
        v16bf b0 = load_fragB(Wic, H2, n0 + l15, k0, khalf);
        acc = WMMA_BF16(a0, b0, acc);
        v16bf a1 = load_fragA(arow1, k0, khalf);
        v16bf b1 = load_fragB(Whh, H2, n0 + l15, k0, khalf);
        acc = WMMA_BF16(a1, b1, acc);
    }
#pragma unroll
    for (int r = 0; r < 8; ++r) {
        const size_t off = (size_t)(m0 + khalf * 8 + r) * H8 + n0 + l15;
        Gd[off] = acc[r] + Xdt[off];
    }
}

// ---------------------------------------------------------------------------
// Elementwise / data-movement kernels
// ---------------------------------------------------------------------------
__global__ void cast_bf16_kernel(bf16_t* __restrict__ dst, const float* __restrict__ src,
                                 int rows, int cols, int src_ld, int src_off)
{
    size_t i = (size_t)blockIdx.x * 256 + threadIdx.x;
    if (i >= (size_t)rows * cols) return;
    int r = (int)(i / cols), c = (int)(i % cols);
    dst[i] = (bf16_t)src[(size_t)r * src_ld + src_off + c];
}

__global__ void gather_emb_kernel(bf16_t* __restrict__ dst, const float* __restrict__ table,
                                  const int* __restrict__ tok, int T)
{
    int t = blockIdx.x, b = blockIdx.y;
    int token = tok[b * T + t];
    const float* srow = table + (size_t)token * EDIM;
    bf16_t* drow = dst + ((size_t)t * BB + b) * EDIM;
    for (int j = threadIdx.x; j < EDIM; j += 256) drow[j] = (bf16_t)srow[j];
}

__global__ void gates_enc_kernel(float* cf, bf16_t* hf, float* cb, bf16_t* hb,
                                 const float* __restrict__ Gf, const float* __restrict__ Gb,
                                 bf16_t* __restrict__ enc_out, int t)
{
    int dir = blockIdx.y;
    int idx = blockIdx.x * 256 + threadIdx.x;
    if (idx >= BB * HDIM) return;
    int b = idx >> 9, j = idx & (HDIM - 1);
    const float* G = dir ? Gb : Gf;
    float* c = dir ? cb : cf;
    bf16_t* h = dir ? hb : hf;
    float ig = sigmoidf_(G[b * H4 + j]);
    float fg = sigmoidf_(G[b * H4 + HDIM + j]);
    float gg = tanhf   (G[b * H4 + 2 * HDIM + j]);
    float og = sigmoidf_(G[b * H4 + 3 * HDIM + j]);
    float cn = fg * c[idx] + ig * gg;
    float hn = og * tanhf(cn);
    c[idx] = cn;
    h[idx] = (bf16_t)hn;
    int tt = dir ? (TSRC - 1 - t) : t;
    enc_out[((size_t)tt * BB + b) * H2 + dir * HDIM + j] = (bf16_t)hn;
}

__global__ void concat_hc_kernel(bf16_t* h_d, float* c_d, const bf16_t* hf, const bf16_t* hb,
                                 const float* cf, const float* cb)
{
    int idx = blockIdx.x * 256 + threadIdx.x;
    if (idx >= BB * HDIM) return;
    int b = idx >> 9, j = idx & (HDIM - 1);
    h_d[b * H2 + j]        = hf[idx];
    h_d[b * H2 + HDIM + j] = hb[idx];
    c_d[b * H2 + j]        = cf[idx];
    c_d[b * H2 + HDIM + j] = cb[idx];
}

// fused attention: score -> softmax -> context vector, one block per batch b
__global__ __launch_bounds__(256) void attn_fused_kernel(
    bf16_t* __restrict__ cvec, const float* __restrict__ Hq, const float* __restrict__ Ea,
    const float* __restrict__ va_, const bf16_t* __restrict__ enc_out)
{
    int b = blockIdx.x;
    __shared__ float sc[TSRC];
    __shared__ float alpha[TSRC];
    const int wave = threadIdx.x >> 5, lane = threadIdx.x & 31;
    const float* hq = Hq + (size_t)b * H2;
    for (int s = wave; s < TSRC; s += 8) {
        const float* ea = Ea + ((size_t)s * BB + b) * H2;
        float sum = 0.0f;
        for (int j = lane; j < H2; j += 32) sum += va_[j] * tanhf(hq[j] + ea[j]);
        for (int off = 16; off > 0; off >>= 1) sum += __shfl_down(sum, off, 32);
        if (lane == 0) sc[s] = sum;
    }
    __syncthreads();
    if (threadIdx.x == 0) {
        float m = -1e30f;
        for (int s = 0; s < TSRC; ++s) m = fmaxf(m, sc[s]);
        float sum = 0.0f;
        for (int s = 0; s < TSRC; ++s) { float e = __expf(sc[s] - m); alpha[s] = e; sum += e; }
        float inv = 1.0f / sum;
        for (int s = 0; s < TSRC; ++s) alpha[s] *= inv;
    }
    __syncthreads();
    for (int j = threadIdx.x; j < H2; j += 256) {
        float acc = 0.0f;
        for (int s = 0; s < TSRC; ++s)
            acc += alpha[s] * (float)enc_out[((size_t)s * BB + b) * H2 + j];
        cvec[(size_t)b * H2 + j] = (bf16_t)acc;
    }
}

__global__ void gates_dec_kernel(float* c_d, bf16_t* h_d, bf16_t* __restrict__ hs,
                                 const float* __restrict__ Gd, int t)
{
    int idx = blockIdx.x * 256 + threadIdx.x;
    if (idx >= BB * H2) return;
    int b = idx >> 10, j = idx & (H2 - 1);
    float ig = sigmoidf_(Gd[b * H8 + j]);
    float fg = sigmoidf_(Gd[b * H8 + H2 + j]);
    float gg = tanhf   (Gd[b * H8 + 2 * H2 + j]);
    float og = sigmoidf_(Gd[b * H8 + 3 * H2 + j]);
    float cn = fg * c_d[idx] + ig * gg;
    float hn = og * tanhf(cn);
    c_d[idx] = cn;
    h_d[idx] = (bf16_t)hn;
    hs[((size_t)b * TTGT + t) * H2 + j] = (bf16_t)hn;
}

__global__ void log_softmax_kernel(float* __restrict__ out)
{
    int r = blockIdx.x;
    float* row = out + (size_t)r * VOUT;
    __shared__ float red[256];
    float m = -1e30f;
    for (int j = threadIdx.x; j < VOUT; j += 256) m = fmaxf(m, row[j]);
    red[threadIdx.x] = m;
    __syncthreads();
    for (int st = 128; st > 0; st >>= 1) {
        if (threadIdx.x < st) red[threadIdx.x] = fmaxf(red[threadIdx.x], red[threadIdx.x + st]);
        __syncthreads();
    }
    m = red[0];
    __syncthreads();
    float s = 0.0f;
    for (int j = threadIdx.x; j < VOUT; j += 256) s += __expf(row[j] - m);
    red[threadIdx.x] = s;
    __syncthreads();
    for (int st = 128; st > 0; st >>= 1) {
        if (threadIdx.x < st) red[threadIdx.x] += red[threadIdx.x + st];
        __syncthreads();
    }
    float lse = m + __logf(red[0]);
    for (int j = threadIdx.x; j < VOUT; j += 256) row[j] -= lse;
}

// ---------------------------------------------------------------------------
// Host side
// ---------------------------------------------------------------------------
static inline void* ws_alloc(char*& p, size_t bytes)
{
    void* r = (void*)p;
    p += (bytes + 255) & ~(size_t)255;
    return r;
}

static void launch_big(hipStream_t s, float* C, const bf16_t* A, const bf16_t* W,
                       const float* addC, const float* bias,
                       int M, int N, int K, int lda, int ldw, int ldc, int relu)
{
    dim3 grid(N / NROWS, M / 128);
    gemm_big_kernel<<<grid, 256, 0, s>>>(C, A, W, addC, bias, M, N, K, lda, ldw, ldc, relu);
}

static void launch_cast(hipStream_t s, bf16_t* dst, const float* src,
                        int rows, int cols, int ld, int off)
{
    size_t total = (size_t)rows * cols;
    cast_bf16_kernel<<<(unsigned)((total + 255) / 256), 256, 0, s>>>(dst, src, rows, cols, ld, off);
}

extern "C" void kernel_launch(void* const* d_in, const int* in_sizes, int n_in,
                              void* d_out, int out_size, void* d_ws, size_t ws_size,
                              hipStream_t stream)
{
    (void)in_sizes; (void)n_in; (void)out_size;

    const int*   inp     = (const int*)  d_in[0];
    const int*   tar     = (const int*)  d_in[1];
    const float* enc_emb = (const float*)d_in[2];
    const float* dec_emb = (const float*)d_in[3];
    const float* Wih_f   = (const float*)d_in[4];
    const float* Whh_f   = (const float*)d_in[5];
    const float* b_f     = (const float*)d_in[6];
    const float* Wih_b   = (const float*)d_in[7];
    const float* Whh_b   = (const float*)d_in[8];
    const float* b_b     = (const float*)d_in[9];
    const float* Wa      = (const float*)d_in[10];
    const float* va      = (const float*)d_in[11];
    const float* Wih_d   = (const float*)d_in[12];
    const float* Whh_d   = (const float*)d_in[13];
    const float* b_d     = (const float*)d_in[14];
    const float* Wout    = (const float*)d_in[15];
    const float* bout    = (const float*)d_in[16];
    float* out = (float*)d_out;

    char* p = (char*)d_ws;
    bf16_t* enc_emb_bf = (bf16_t*)ws_alloc(p, (size_t)TSRC * BB * EDIM * 2);
    bf16_t* dec_emb_bf = (bf16_t*)ws_alloc(p, (size_t)TTGT * BB * EDIM * 2);
    bf16_t* Wih_f_bf   = (bf16_t*)ws_alloc(p, (size_t)H4 * EDIM * 2);
    bf16_t* Whh_f_bf   = (bf16_t*)ws_alloc(p, (size_t)H4 * HDIM * 2);
    bf16_t* Wih_b_bf   = (bf16_t*)ws_alloc(p, (size_t)H4 * EDIM * 2);
    bf16_t* Whh_b_bf   = (bf16_t*)ws_alloc(p, (size_t)H4 * HDIM * 2);
    bf16_t* Wa_h_bf    = (bf16_t*)ws_alloc(p, (size_t)H2 * H2 * 2);
    bf16_t* Wa_e_bf    = (bf16_t*)ws_alloc(p, (size_t)H2 * H2 * 2);
    bf16_t* Wih_dx_bf  = (bf16_t*)ws_alloc(p, (size_t)H8 * EDIM * 2);
    bf16_t* Wih_dc_bf  = (bf16_t*)ws_alloc(p, (size_t)H8 * H2 * 2);
    bf16_t* Whh_d_bf   = (bf16_t*)ws_alloc(p, (size_t)H8 * H2 * 2);
    bf16_t* Wout_bf    = (bf16_t*)ws_alloc(p, (size_t)VOUT * H2 * 2);
    float*  Xf         = (float*) ws_alloc(p, (size_t)TSRC * BB * H4 * 4);
    float*  Xb         = (float*) ws_alloc(p, (size_t)TSRC * BB * H4 * 4);
    float*  Xd         = (float*) ws_alloc(p, (size_t)TTGT * BB * H8 * 4);
    bf16_t* enc_out_bf = (bf16_t*)ws_alloc(p, (size_t)TSRC * BB * H2 * 2);
    float*  Ea         = (float*) ws_alloc(p, (size_t)TSRC * BB * H2 * 4);
    bf16_t* hs_bf      = (bf16_t*)ws_alloc(p, (size_t)BB * TTGT * H2 * 2);
    float*  Gf         = (float*) ws_alloc(p, (size_t)BB * H4 * 4);
    float*  Gb         = (float*) ws_alloc(p, (size_t)BB * H4 * 4);
    float*  Gd         = (float*) ws_alloc(p, (size_t)BB * H8 * 4);
    float*  cf         = (float*) ws_alloc(p, (size_t)BB * HDIM * 4);
    float*  cb         = (float*) ws_alloc(p, (size_t)BB * HDIM * 4);
    bf16_t* hf_bf      = (bf16_t*)ws_alloc(p, (size_t)BB * HDIM * 2);
    bf16_t* hb_bf      = (bf16_t*)ws_alloc(p, (size_t)BB * HDIM * 2);
    float*  c_d        = (float*) ws_alloc(p, (size_t)BB * H2 * 4);
    bf16_t* h_d_bf     = (bf16_t*)ws_alloc(p, (size_t)BB * H2 * 2);
    bf16_t* cvec_bf    = (bf16_t*)ws_alloc(p, (size_t)BB * H2 * 2);
    float*  Hq         = (float*) ws_alloc(p, (size_t)BB * H2 * 4);
    if ((size_t)(p - (char*)d_ws) > ws_size) return;

    // weight staging fp32 -> bf16 (with strided splits of Wa / Wih_d)
    launch_cast(stream, Wih_f_bf, Wih_f, H4, EDIM, EDIM, 0);
    launch_cast(stream, Whh_f_bf, Whh_f, H4, HDIM, HDIM, 0);
    launch_cast(stream, Wih_b_bf, Wih_b, H4, EDIM, EDIM, 0);
    launch_cast(stream, Whh_b_bf, Whh_b, H4, HDIM, HDIM, 0);
    launch_cast(stream, Wa_h_bf, Wa, H2, H2, H4, 0);
    launch_cast(stream, Wa_e_bf, Wa, H2, H2, H4, H2);
    launch_cast(stream, Wih_dx_bf, Wih_d, H8, EDIM, EDIM + H2, 0);
    launch_cast(stream, Wih_dc_bf, Wih_d, H8, H2, EDIM + H2, EDIM);
    launch_cast(stream, Whh_d_bf, Whh_d, H8, H2, H2, 0);
    launch_cast(stream, Wout_bf, Wout, VOUT, H2, H2, 0);

    {
        dim3 g(TSRC, BB);
        gather_emb_kernel<<<g, 256, 0, stream>>>(enc_emb_bf, enc_emb, inp, TSRC);
        gather_emb_kernel<<<g, 256, 0, stream>>>(dec_emb_bf, dec_emb, tar, TTGT);
    }

    hipMemsetAsync(cf, 0, (size_t)BB * HDIM * 4, stream);
    hipMemsetAsync(cb, 0, (size_t)BB * HDIM * 4, stream);
    hipMemsetAsync(hf_bf, 0, (size_t)BB * HDIM * 2, stream);
    hipMemsetAsync(hb_bf, 0, (size_t)BB * HDIM * 2, stream);

    // hoisted input-side GEMMs
    launch_big(stream, Xf, enc_emb_bf, Wih_f_bf, nullptr, b_f,
               TSRC * BB, H4, EDIM, EDIM, EDIM, H4, 0);
    launch_big(stream, Xb, enc_emb_bf, Wih_b_bf, nullptr, b_b,
               TSRC * BB, H4, EDIM, EDIM, EDIM, H4, 0);
    launch_big(stream, Xd, dec_emb_bf, Wih_dx_bf, nullptr, b_d,
               TTGT * BB, H8, EDIM, EDIM, EDIM, H8, 0);

    // encoder scan: one fused fwd+bwd GEMM launch + one gates launch per step
    for (int t = 0; t < TSRC; ++t) {
        dim3 gg(16, 2, 2);      // N=2048 -> 128 tiles / 8 waves, M=32 -> 2, dirs
        enc_step_kernel<<<gg, 256, 0, stream>>>(Gf, Gb, hf_bf, hb_bf, Whh_f_bf, Whh_b_bf,
                                                Xf + (size_t)t * BB * H4,
                                                Xb + (size_t)(TSRC - 1 - t) * BB * H4);
        dim3 ge((BB * HDIM + 255) / 256, 2);
        gates_enc_kernel<<<ge, 256, 0, stream>>>(cf, hf_bf, cb, hb_bf, Gf, Gb, enc_out_bf, t);
    }

    concat_hc_kernel<<<(BB * HDIM + 255) / 256, 256, 0, stream>>>(h_d_bf, c_d, hf_bf, hb_bf, cf, cb);
    // hoisted attention term: Ea = enc_out @ Wa_e.T
    launch_big(stream, Ea, enc_out_bf, Wa_e_bf, nullptr, nullptr,
               TSRC * BB, H2, H2, H2, H2, H2, 0);

    // decoder scan: 4 launches per step
    for (int t = 0; t < TTGT; ++t) {
        dim3 gh(8, 2);          // N=1024 -> 64 tiles / 8 waves, M=32 -> 2
        gemm_small_kernel<<<gh, 256, 0, stream>>>(Hq, h_d_bf, Wa_h_bf, BB, H2, H2, H2, H2, H2);
        attn_fused_kernel<<<BB, 256, 0, stream>>>(cvec_bf, Hq, Ea, va, enc_out_bf);
        dim3 gd(32, 2);         // N=4096 -> 256 tiles / 8 waves, M=32 -> 2
        dec_step_kernel<<<gd, 256, 0, stream>>>(Gd, cvec_bf, Wih_dc_bf, h_d_bf, Whh_d_bf,
                                                Xd + (size_t)t * BB * H8);
        gates_dec_kernel<<<(BB * H2 + 255) / 256, 256, 0, stream>>>(c_d, h_d_bf, hs_bf, Gd, t);
    }

    // output projection (dominant GEMM) + relu into d_out, then log_softmax
    launch_big(stream, out, hs_bf, Wout_bf, nullptr, bout,
               BB * TTGT, VOUT, H2, H2, H2, VOUT, 1);
    log_softmax_kernel<<<BB * TTGT, 256, 0, stream>>>(out);
}
